// AttnDecoder_87230785782556
// MI455X (gfx1250) — matrix operand order repaired and checked by reading, not compiled, not run on hardware
//
#include <hip/hip_runtime.h>

// ---------------------------------------------------------------------------
// MHA block on gfx1250 (MI455X): all GEMMs via v_wmma_f32_16x16x32_bf16,
// flash-attention K/V staging via CDNA5 async LDS DMA
// (global_load_async_to_lds_b128 + s_wait_asynccnt).
//   B=2, Lq=Lkv=2048, E=1024, H=16, C=64.
// ---------------------------------------------------------------------------

typedef __attribute__((ext_vector_type(16))) __bf16 v16bf;
typedef __attribute__((ext_vector_type(8)))  __bf16 v8bf;
typedef __attribute__((ext_vector_type(4)))  __bf16 v4bf;
typedef __attribute__((ext_vector_type(8)))  float  v8f;

#define NB   2
#define LQ   2048
#define LKV  2048
#define EMB  1024
#define NH   16
#define HD   64

static __device__ inline v8f wmma_bf16(v16bf a, v16bf b, v8f c) {
  return __builtin_amdgcn_wmma_f32_16x16x32_bf16(false, a, false, b,
                                                 (short)0, c, false, false);
}

// A-fragment (M=16 x K=32, row-major source, row stride ld elements).
// Lanes 0-15 -> M=lane, K 0..15; lanes 16-31 -> K 16..31.
static __device__ inline v16bf load_frag_a(const __bf16* base, int ld) {
  const int lane = threadIdx.x & 31;
  const __bf16* p = base + (size_t)(lane & 15) * ld + ((lane >> 4) << 4);
  v8bf lo = *(const v8bf*)p;          // 16B
  v8bf hi = *(const v8bf*)(p + 8);    // 16B
  v16bf r;
#pragma unroll
  for (int j = 0; j < 8; ++j) { r[j] = lo[j]; r[8 + j] = hi[j]; }
  return r;
}

// B-fragment (K=32 x N=16) from B^T stored row-major [N][K], row stride ld.
// Lane = column N (lane&15); 16 contiguous K halves at (lane>>4)*16.
static __device__ inline v16bf load_frag_b(const __bf16* baseT, int ld) {
  const int lane = threadIdx.x & 31;
  const __bf16* p = baseT + (size_t)(lane & 15) * ld + ((lane >> 4) << 4);
  return *(const v16bf*)p;            // 32B
}

static __device__ inline float red16_max(float v) {
  v = fmaxf(v, __shfl_xor(v, 1, 32));
  v = fmaxf(v, __shfl_xor(v, 2, 32));
  v = fmaxf(v, __shfl_xor(v, 4, 32));
  v = fmaxf(v, __shfl_xor(v, 8, 32));
  return v;
}
static __device__ inline float red16_sum(float v) {
  v += __shfl_xor(v, 1, 32);
  v += __shfl_xor(v, 2, 32);
  v += __shfl_xor(v, 4, 32);
  v += __shfl_xor(v, 8, 32);
  return v;
}

// CDNA5 async memory->LDS DMA of one 16B chunk (ASYNCcnt tracked, no VGPR
// round trip). ldsaddr = low 32 bits of generic shared pointer (ISA 10.2).
static __device__ inline void async_ld_lds_b128(void* lds, const void* gaddr) {
  unsigned ldsaddr = (unsigned)(unsigned long long)lds;
  asm volatile("global_load_async_to_lds_b128 %0, %1, off"
               :: "v"(ldsaddr), "v"(gaddr) : "memory");
}
static __device__ inline void wait_async0() {
  asm volatile("s_wait_asynccnt 0x0" ::: "memory");
}

// ---------------------------------------------------------------------------
// fp32 -> bf16 elementwise (4 per thread)
// ---------------------------------------------------------------------------
__global__ __launch_bounds__(256) void cvt_f32_bf16(const float* __restrict__ src,
                                                    unsigned short* __restrict__ dst,
                                                    int n4) {
  int i = blockIdx.x * blockDim.x + threadIdx.x;
  if (i < n4) {
    float4 v = ((const float4*)src)[i];
    v4bf o;
    o[0] = (__bf16)v.x; o[1] = (__bf16)v.y; o[2] = (__bf16)v.z; o[3] = (__bf16)v.w;
    *(v4bf*)((__bf16*)dst + (size_t)i * 4) = o;
  }
}

// ---------------------------------------------------------------------------
// W (1024x1024 f32) -> Wt (1024x1024 bf16, transposed) via 32x32 LDS tile
// ---------------------------------------------------------------------------
__global__ __launch_bounds__(256) void transpose_cvt(const float* __restrict__ W,
                                                     unsigned short* __restrict__ Wt) {
  __shared__ float tile[32][33];
  const int tx = threadIdx.x & 31, ty = threadIdx.x >> 5;   // ty: 0..7
  const int bx = blockIdx.x, by = blockIdx.y;
#pragma unroll
  for (int k = 0; k < 4; ++k)
    tile[ty + 8 * k][tx] = W[(size_t)(by * 32 + ty + 8 * k) * EMB + bx * 32 + tx];
  __syncthreads();
  __bf16* out = (__bf16*)Wt;
#pragma unroll
  for (int k = 0; k < 4; ++k)
    out[(size_t)(bx * 32 + ty + 8 * k) * EMB + by * 32 + tx] =
        (__bf16)tile[tx][ty + 8 * k];
}

// ---------------------------------------------------------------------------
// WMMA GEMM: C(4096x1024) = A(4096x1024,bf16) * B(1024x1024), B given as
// B^T bf16 (N-major). Block tile 128x128, 8 waves, wave tile 64x32.
// K-loop ping-pong double-buffered in registers (prefetch next 32-K frags).
// mode 0: Qh[bh][l][c] bf16 (scaled)   mode 1: Kh[bh][l][c] bf16
// mode 2: Vt[bh][c][l]  bf16           mode 3: f32 plain [row][e]
// ---------------------------------------------------------------------------
__global__ __launch_bounds__(256) void gemm_wmma_bf16(
    const unsigned short* __restrict__ A, const unsigned short* __restrict__ Bt,
    unsigned short* __restrict__ obf, float* __restrict__ of32,
    int mode, float scale) {
  const int wave = threadIdx.x >> 5;
  const int lane = threadIdx.x & 31;
  const int wm = wave >> 2;            // 0..1
  const int wn = wave & 3;             // 0..3
  const int m0 = blockIdx.x * 128 + wm * 64;
  const int n0 = blockIdx.y * 128 + wn * 32;
  const __bf16* Ab  = (const __bf16*)A;
  const __bf16* Btb = (const __bf16*)Bt;

  v8f zero = {0.f, 0.f, 0.f, 0.f, 0.f, 0.f, 0.f, 0.f};
  v8f acc[4][2];
#pragma unroll
  for (int i = 0; i < 4; ++i)
#pragma unroll
    for (int j = 0; j < 2; ++j) acc[i][j] = zero;

  v16bf a0[4], b0[2], a1[4], b1[2];
  auto load_set = [&](v16bf* af, v16bf* bfr, int kb) {
#pragma unroll
    for (int i = 0; i < 4; ++i)
      af[i] = load_frag_a(Ab + (size_t)(m0 + 16 * i) * EMB + kb, EMB);
#pragma unroll
    for (int j = 0; j < 2; ++j)
      bfr[j] = load_frag_b(Btb + (size_t)(n0 + 16 * j) * EMB + kb, EMB);
  };
  auto do_wmma = [&](v16bf* af, v16bf* bfr) {
#pragma unroll
    for (int i = 0; i < 4; ++i)
#pragma unroll
      for (int j = 0; j < 2; ++j)
        acc[i][j] = wmma_bf16(af[i], bfr[j], acc[i][j]);
  };

  load_set(a0, b0, 0);
  for (int kb = 0; kb < EMB; kb += 64) {
    load_set(a1, b1, kb + 32);           // prefetch while buf0 computes
    do_wmma(a0, b0);
    if (kb + 64 < EMB) load_set(a0, b0, kb + 64);
    do_wmma(a1, b1);
  }

  const int nl = lane & 15, half = lane >> 4;
  __bf16* ob = (__bf16*)obf;
#pragma unroll
  for (int i = 0; i < 4; ++i) {
#pragma unroll
    for (int j = 0; j < 2; ++j) {
#pragma unroll
      for (int r = 0; r < 8; ++r) {
        const int row = m0 + 16 * i + r + half * 8;   // 0..4095
        const int e   = n0 + 16 * j + nl;             // 0..1023
        const float v = acc[i][j][r] * scale;
        if (mode == 3) {
          of32[(size_t)row * EMB + e] = v;
        } else {
          const int b = row >> 11, l = row & 2047;
          const int h = e & 15, c = e >> 4;
          const int bh = b * NH + h;
          if (mode == 2)
            ob[((size_t)bh * HD + c) * LKV + l] = (__bf16)v;     // Vt[bh][c][l]
          else
            ob[((size_t)bh * LKV + l) * HD + c] = (__bf16)v;     // Qh/Kh[bh][l][c]
        }
      }
    }
  }
}

// ---------------------------------------------------------------------------
// Flash attention. Grid: (LQ/128, B*H). Block: 256 threads = 8 waves.
// Each wave owns 16 q-rows; kv tiled by 64 with K/V staged in LDS via
// async DMA (ASYNCcnt). Q pre-scaled by 0.125/sqrt(2)*log2(e) -> exp2 domain.
// ---------------------------------------------------------------------------
__global__ __launch_bounds__(256) void flash_attn(
    const unsigned short* __restrict__ Qh, const unsigned short* __restrict__ Kh,
    const unsigned short* __restrict__ Vt, unsigned short* __restrict__ Om) {
  __shared__ __align__(16) __bf16 KT[64 * 64];       // [kv][c]
  __shared__ __align__(16) __bf16 VT[64 * 64];       // [c][kv]
  __shared__ __align__(16) __bf16 Psh[8][16 * 64];   // per-wave P tile

  const int tid = threadIdx.x;
  const int w = tid >> 5;
  const int lane = tid & 31;
  const int nl = lane & 15, half = lane >> 4;
  const int bh = blockIdx.y;
  const int b = bh >> 4, h = bh & 15;
  const int qr = blockIdx.x * 128 + w * 16;

  const __bf16* Qb = (const __bf16*)Qh + ((size_t)bh * LQ + qr) * HD;
  const __bf16* Kb = (const __bf16*)Kh + (size_t)bh * LKV * HD;
  const __bf16* Vb = (const __bf16*)Vt + (size_t)bh * HD * LKV;

  const v16bf qf0 = load_frag_a(Qb, HD);
  const v16bf qf1 = load_frag_a(Qb + 32, HD);

  v8f zero = {0.f, 0.f, 0.f, 0.f, 0.f, 0.f, 0.f, 0.f};
  v8f oacc[4];
#pragma unroll
  for (int j = 0; j < 4; ++j) oacc[j] = zero;
  float mrow[8], lrow[8];
#pragma unroll
  for (int r = 0; r < 8; ++r) { mrow[r] = -1e30f; lrow[r] = 0.f; }

  for (int kv = 0; kv < LKV; kv += 64) {
    // --- async DMA K tile (contiguous 8KB) and V tile (64 rows x 128B) ---
    {
      const __bf16* srcK = Kb + (size_t)kv * HD;
#pragma unroll
      for (int g = tid; g < 512; g += 256)
        async_ld_lds_b128(KT + g * 8, srcK + g * 8);
#pragma unroll
      for (int g = tid; g < 512; g += 256) {
        const int cc = g >> 3, wi = g & 7;
        async_ld_lds_b128(VT + g * 8, Vb + (size_t)cc * LKV + kv + wi * 8);
      }
      wait_async0();
    }
    __syncthreads();

    // --- scores S (16x64) = Q * K^T ---
    v8f sblk[4];
#pragma unroll
    for (int ns = 0; ns < 4; ++ns) {
      v8f s = zero;
      s = wmma_bf16(qf0, load_frag_b(&KT[ns * 16 * 64 + 0], 64), s);
      s = wmma_bf16(qf1, load_frag_b(&KT[ns * 16 * 64 + 32], 64), s);
      sblk[ns] = s;
    }

    // --- online softmax (exp2 domain) ---
    float mnew[8], alpha[8], rs[8];
#pragma unroll
    for (int r = 0; r < 8; ++r) {
      float t = fmaxf(fmaxf(sblk[0][r], sblk[1][r]), fmaxf(sblk[2][r], sblk[3][r]));
      t = red16_max(t);
      const float mn = fmaxf(mrow[r], t);
      alpha[r] = exp2f(mrow[r] - mn);
      mnew[r] = mn;
      rs[r] = 0.f;
    }
#pragma unroll
    for (int ns = 0; ns < 4; ++ns) {
#pragma unroll
      for (int r = 0; r < 8; ++r) {
        const float pe = exp2f(sblk[ns][r] - mnew[r]);
        rs[r] += pe;
        Psh[w][(r + half * 8) * 64 + ns * 16 + nl] = (__bf16)pe;
      }
    }
#pragma unroll
    for (int r = 0; r < 8; ++r) {
      lrow[r] = lrow[r] * alpha[r] + red16_sum(rs[r]);
      mrow[r] = mnew[r];
    }
#pragma unroll
    for (int j = 0; j < 4; ++j)
#pragma unroll
      for (int r = 0; r < 8; ++r) oacc[j][r] *= alpha[r];

    // --- O += P * V ---
#pragma unroll
    for (int kk = 0; kk < 64; kk += 32) {
      const v16bf pa = load_frag_a(&Psh[w][kk], 64);
#pragma unroll
      for (int j = 0; j < 4; ++j)
        oacc[j] = wmma_bf16(pa, load_frag_b(&VT[j * 16 * 64 + kk], 64), oacc[j]);
    }
    __syncthreads();
  }

  // --- normalize + merged-head bf16 output: e = c*16 + h ---
  float inv[8];
#pragma unroll
  for (int r = 0; r < 8; ++r) inv[r] = 1.0f / lrow[r];
  __bf16* ob = (__bf16*)Om;
#pragma unroll
  for (int j = 0; j < 4; ++j) {
#pragma unroll
    for (int r = 0; r < 8; ++r) {
      const int q = qr + r + half * 8;
      const int c = j * 16 + nl;
      const int e = c * NH + h;
      ob[((size_t)b * LQ + q) * EMB + e] = (__bf16)(oacc[j][r] * inv[r]);
    }
  }
}

// ---------------------------------------------------------------------------
extern "C" void kernel_launch(void* const* d_in, const int* in_sizes, int n_in,
                              void* d_out, int out_size, void* d_ws, size_t ws_size,
                              hipStream_t stream) {
  (void)in_sizes; (void)n_in; (void)out_size; (void)ws_size;
  const float* q  = (const float*)d_in[0];
  const float* k  = (const float*)d_in[1];
  const float* v  = (const float*)d_in[2];
  const float* Wq = (const float*)d_in[3];
  const float* Wk = (const float*)d_in[4];
  const float* Wv = (const float*)d_in[5];
  const float* Wo = (const float*)d_in[6];

  const size_t MB = 1ull << 20;
  char* ws = (char*)d_ws;
  unsigned short* qbf    = (unsigned short*)(ws + 0 * MB);   // 8 MB
  unsigned short* kbf    = (unsigned short*)(ws + 8 * MB);   // 8 MB
  unsigned short* vbf    = (unsigned short*)(ws + 16 * MB);  // 8 MB
  unsigned short* WqT    = (unsigned short*)(ws + 24 * MB);  // 2 MB
  unsigned short* WkT    = (unsigned short*)(ws + 26 * MB);  // 2 MB
  unsigned short* WvT    = (unsigned short*)(ws + 28 * MB);  // 2 MB
  unsigned short* WoT    = (unsigned short*)(ws + 30 * MB);  // 2 MB
  unsigned short* Qh     = (unsigned short*)(ws + 32 * MB);  // 8 MB
  unsigned short* Kh     = (unsigned short*)(ws + 40 * MB);  // 8 MB
  unsigned short* Vt     = (unsigned short*)(ws + 48 * MB);  // 8 MB
  unsigned short* merged = (unsigned short*)(ws + 56 * MB);  // 8 MB

  const int NQKV4 = (NB * LQ * EMB) / 4;                      // 1048576
  cvt_f32_bf16<<<NQKV4 / 256, 256, 0, stream>>>(q, qbf, NQKV4);
  cvt_f32_bf16<<<NQKV4 / 256, 256, 0, stream>>>(k, kbf, NQKV4);
  cvt_f32_bf16<<<NQKV4 / 256, 256, 0, stream>>>(v, vbf, NQKV4);

  dim3 tg(EMB / 32, EMB / 32);
  transpose_cvt<<<tg, 256, 0, stream>>>(Wq, WqT);
  transpose_cvt<<<tg, 256, 0, stream>>>(Wk, WkT);
  transpose_cvt<<<tg, 256, 0, stream>>>(Wv, WvT);
  transpose_cvt<<<tg, 256, 0, stream>>>(Wo, WoT);

  // qk_scale * 1/sqrt(batch) * log2(e) folded into Q projection
  const float QSCALE = 0.125f * 0.70710678118654752f * 1.4426950408889634f;
  dim3 gg((NB * LQ) / 128, EMB / 128);                        // (32, 8)
  gemm_wmma_bf16<<<gg, 256, 0, stream>>>(qbf, WqT, Qh, nullptr, 0, QSCALE);
  gemm_wmma_bf16<<<gg, 256, 0, stream>>>(kbf, WkT, Kh, nullptr, 1, 1.0f);
  gemm_wmma_bf16<<<gg, 256, 0, stream>>>(vbf, WvT, Vt, nullptr, 2, 1.0f);

  dim3 fg(LQ / 128, NB * NH);                                 // (16, 32)
  flash_attn<<<fg, 256, 0, stream>>>(Qh, Kh, Vt, merged);

  gemm_wmma_bf16<<<gg, 256, 0, stream>>>(merged, WoT, nullptr, (float*)d_out, 3, 1.0f);
}
